// Attention_50139448213963
// MI455X (gfx1250) — compile-verified, matrix operand order checked by reading, not run on hardware
//
#include <hip/hip_runtime.h>
#include <hip/hip_bf16.h>

// ---------------------------------------------------------------------------
// Attention: f16 pre-convert -> qkv proj (WMMA) -> per-head LN + RoPE ->
// flash attention (WMMA, async LDS staging) -> out proj (WMMA).
// All matrix math on v_wmma_f32_16x16x32_f16 (wave32, gfx1250).
// ---------------------------------------------------------------------------

#define DIMX 1024
#define NH   8
#define HD   128
#define BB   2
#define NN   2048
#define RTOT (BB * NN)      // 4096 token rows
#define C3   (3 * DIMX)     // 3072

typedef __attribute__((ext_vector_type(16))) _Float16 v16h;
typedef __attribute__((ext_vector_type(8)))  float    v8f;
typedef int v4i __attribute__((vector_size(16)));   // matches async-LDS builtin param

#define AS1 __attribute__((address_space(1)))
#define AS3 __attribute__((address_space(3)))

#if defined(__has_builtin)
#if __has_builtin(__builtin_amdgcn_global_load_async_to_lds_b128)
#define HAVE_ASYNC_LDS 1
#endif
#endif
#ifndef HAVE_ASYNC_LDS
#define HAVE_ASYNC_LDS 0
#endif

__device__ __forceinline__ v8f wmma_f16(v16h a, v16h b, v8f c) {
    return __builtin_amdgcn_wmma_f32_16x16x32_f16(
        /*neg_a=*/false, a, /*neg_b=*/false, b,
        /*c_mod=*/(short)0, c, /*reuse_a=*/false, /*reuse_b=*/false);
}

__device__ __forceinline__ v8f zero8() {
    v8f z = {0.f, 0.f, 0.f, 0.f, 0.f, 0.f, 0.f, 0.f};
    return z;
}

// 16-byte global -> LDS copy: async on gfx1250 when the builtin exists.
__device__ __forceinline__ void cp16_g2l(const _Float16* __restrict__ g,
                                         _Float16* __restrict__ l) {
#if HAVE_ASYNC_LDS
    __builtin_amdgcn_global_load_async_to_lds_b128(
        (AS1 v4i*)(uintptr_t)g,
        (AS3 v4i*)(unsigned)(uintptr_t)l, 0, 0);
#else
    *(uint4*)l = *(const uint4*)g;
#endif
}

__device__ __forceinline__ void drain_async() {
#if HAVE_ASYNC_LDS
    asm volatile("s_wait_asynccnt 0x0" ::: "memory");
#endif
}

// ---------------------------------------------------------------------------
// Kernel 0: fp32 -> f16 conversion pass (x, Wqkv_w, out_w), 4 elems/thread.
// ---------------------------------------------------------------------------
__global__ __launch_bounds__(256) void cvt_f16_k(const float* __restrict__ s,
                                                 _Float16* __restrict__ d) {
    const int i = (blockIdx.x * 256 + threadIdx.x) * 4;
    const float4 v = *(const float4*)(s + i);
    d[i + 0] = (_Float16)v.x;
    d[i + 1] = (_Float16)v.y;
    d[i + 2] = (_Float16)v.z;
    d[i + 3] = (_Float16)v.w;
}

// ---------------------------------------------------------------------------
// Kernel 1: qkv = x @ Wqkv^T + b   (f16 operands, fp32 out to workspace)
// WG: 8 waves as 2x4 -> 64 rows x 256 cols. Wave tile 32x64.
// grid = (4096/64, 3072/256) = (64, 12)
// ---------------------------------------------------------------------------
__global__ __launch_bounds__(256) void qkv_gemm_k(
    const _Float16* __restrict__ xh, const _Float16* __restrict__ wh,
    const float* __restrict__ bias, float* __restrict__ out)
{
    const int wave = threadIdx.x >> 5, lane = threadIdx.x & 31;
    const int wr = wave >> 2, wc = wave & 3;
    const int row0 = blockIdx.x * 64 + wr * 32;
    const int col0 = blockIdx.y * 256 + wc * 64;
    const int l16 = lane & 15, lh = lane >> 4;

    v8f acc[2][4];
#pragma unroll
    for (int i = 0; i < 2; ++i)
#pragma unroll
        for (int j = 0; j < 4; ++j) acc[i][j] = zero8();

    for (int k = 0; k < DIMX; k += 32) {
        v16h a[2], bf[4];
#pragma unroll
        for (int rt = 0; rt < 2; ++rt)
            a[rt] = *(const v16h*)(xh + (size_t)(row0 + rt * 16 + l16) * DIMX + k + lh * 16);
#pragma unroll
        for (int ct = 0; ct < 4; ++ct)
            bf[ct] = *(const v16h*)(wh + (size_t)(col0 + ct * 16 + l16) * DIMX + k + lh * 16);
#pragma unroll
        for (int rt = 0; rt < 2; ++rt)
#pragma unroll
            for (int ct = 0; ct < 4; ++ct)
                acc[rt][ct] = wmma_f16(a[rt], bf[ct], acc[rt][ct]);
    }

#pragma unroll
    for (int rt = 0; rt < 2; ++rt)
#pragma unroll
        for (int ct = 0; ct < 4; ++ct) {
            const int col = col0 + ct * 16 + l16;
            const float bz = bias[col];
#pragma unroll
            for (int r = 0; r < 8; ++r) {
                const int row = row0 + rt * 16 + lh * 8 + r;
                out[(size_t)row * C3 + col] = acc[rt][ct][r] + bz;
            }
        }
}

// ---------------------------------------------------------------------------
// Kernel 2: per-head LayerNorm on q,k + interleaved RoPE; pack q/k to f16
// [B,H,N,D] and v to f16 TRANSPOSED [B,H,D,N] (for async V^T staging).
// One wave per (b,n,h). grid = B*N*H / 8 = 4096 blocks of 256.
// ---------------------------------------------------------------------------
__device__ __forceinline__ void wave_stats(float s, float ss, float& mu, float& rstd) {
#pragma unroll
    for (int m = 16; m >= 1; m >>= 1) {
        s  += __shfl_xor(s,  m, 32);
        ss += __shfl_xor(ss, m, 32);
    }
    mu = s * (1.f / HD);
    const float var = ss * (1.f / HD) - mu * mu;
    rstd = rsqrtf(var + 1e-6f);
}

__global__ __launch_bounds__(256) void lnrope_pack_k(
    const float* __restrict__ qkv,
    const float* __restrict__ qg, const float* __restrict__ qb,
    const float* __restrict__ kg, const float* __restrict__ kb,
    const float* __restrict__ rc, const float* __restrict__ rs,
    _Float16* __restrict__ qh, _Float16* __restrict__ kh, _Float16* __restrict__ vt)
{
    const int wave = threadIdx.x >> 5, lane = threadIdx.x & 31;
    const int idx = blockIdx.x * 8 + wave;   // 0..32767
    const int h  = idx & (NH - 1);
    const int ng = idx >> 3;                 // b*N + n
    const int n  = ng & (NN - 1);
    const int b  = ng >> 11;
    const int d0 = lane * 4;

    const float* rowp = qkv + (size_t)ng * C3;
    const float4 q4 = *(const float4*)(rowp + (0 * NH + h) * HD + d0);
    const float4 k4 = *(const float4*)(rowp + (1 * NH + h) * HD + d0);
    const float4 v4 = *(const float4*)(rowp + (2 * NH + h) * HD + d0);

    float muq, rq, muk, rk;
    wave_stats(q4.x + q4.y + q4.z + q4.w,
               q4.x*q4.x + q4.y*q4.y + q4.z*q4.z + q4.w*q4.w, muq, rq);
    wave_stats(k4.x + k4.y + k4.z + k4.w,
               k4.x*k4.x + k4.y*k4.y + k4.z*k4.z + k4.w*k4.w, muk, rk);

    float qn[4] = {q4.x, q4.y, q4.z, q4.w};
    float kn[4] = {k4.x, k4.y, k4.z, k4.w};
#pragma unroll
    for (int i = 0; i < 4; ++i) {
        qn[i] = (qn[i] - muq) * rq * qg[d0 + i] + qb[d0 + i];
        kn[i] = (kn[i] - muk) * rk * kg[d0 + i] + kb[d0 + i];
    }

    const float* cp = rc + (size_t)n * HD + d0;
    const float* sp = rs + (size_t)n * HD + d0;
    float qo[4], ko[4];
#pragma unroll
    for (int p = 0; p < 2; ++p) {      // two interleaved pairs per lane
        const float c0 = cp[2*p], c1 = cp[2*p + 1];
        const float s0 = sp[2*p], s1 = sp[2*p + 1];
        qo[2*p]     = qn[2*p]     * c0 - qn[2*p + 1] * s0;
        qo[2*p + 1] = qn[2*p + 1] * c1 + qn[2*p]     * s1;
        ko[2*p]     = kn[2*p]     * c0 - kn[2*p + 1] * s0;
        ko[2*p + 1] = kn[2*p + 1] * c1 + kn[2*p]     * s1;
    }

    const size_t qkbase = (((size_t)b * NH + h) * NN + n) * HD + d0;
    const size_t vtbase = (((size_t)b * NH + h) * HD + d0) * NN + n;
    const float vv[4] = {v4.x, v4.y, v4.z, v4.w};
#pragma unroll
    for (int i = 0; i < 4; ++i) {
        qh[qkbase + i] = (_Float16)qo[i];
        kh[qkbase + i] = (_Float16)ko[i];
        vt[vtbase + (size_t)i * NN] = (_Float16)vv[i];
    }
}

// ---------------------------------------------------------------------------
// Kernel 3: flash attention per (b,h). 8 waves x 16 query rows = 128 q/WG.
// K tile [64][128] and V^T tile [128][64] staged to LDS with async b128
// copies (ASYNCcnt-drained, then barrier). Online softmax; P re-laid-out
// via per-wave LDS tile for the PV WMMAs. grid = (B*H, N/128) = (16, 16)
// ---------------------------------------------------------------------------
__global__ __launch_bounds__(256) void attn_k(
    const _Float16* __restrict__ qh, const _Float16* __restrict__ kh,
    const _Float16* __restrict__ vt, _Float16* __restrict__ oh)
{
    __shared__ _Float16 kbuf[64 * 128];        // [j][d]
    __shared__ _Float16 vbuf[128 * 64];        // [d][j]
    __shared__ _Float16 pbuf[8][16 * 64];      // per-wave P tile [row][j]

    const int bh = blockIdx.x;
    const int b = bh >> 3, h = bh & 7;
    const int wave = threadIdx.x >> 5, lane = threadIdx.x & 31;
    const int l16 = lane & 15, lh = lane >> 4;
    const int q0 = blockIdx.y * 128 + wave * 16;

    const _Float16* qp  = qh + ((size_t)bh * NN + q0) * HD;
    const _Float16* kp  = kh + (size_t)bh * NN * HD;
    const _Float16* vtp = vt + (size_t)bh * HD * NN;

    // hoist q fragments: 16 rows x full D=128 (4 k-steps of 32)
    v16h qf[4];
#pragma unroll
    for (int dk = 0; dk < 4; ++dk)
        qf[dk] = *(const v16h*)(qp + (size_t)l16 * HD + dk * 32 + lh * 16);

    v8f o[8];
#pragma unroll
    for (int dt = 0; dt < 8; ++dt) o[dt] = zero8();
    float m[8], l[8];
#pragma unroll
    for (int r = 0; r < 8; ++r) { m[r] = -1e30f; l[r] = 0.f; }

    const float sscale = 0.08838834764831845f;   // 1/sqrt(128)
    const float E2 = 1.4426950408889634f;        // log2(e)

    for (int kt = 0; kt < NN; kt += 64) {
        __syncthreads();   // previous tile fully consumed before restage
        {   // stage K tile and V^T tile: 1024 16B chunks each, 4 per thread
#pragma unroll
            for (int i = 0; i < 4; ++i) {
                const int c  = i * 256 + threadIdx.x;   // 0..1023
                cp16_g2l(kp + (size_t)kt * HD + c * 8, &kbuf[c * 8]);
                const int d  = c >> 3;                  // V^T row 0..127
                const int jc = (c & 7) * 8;             // 8-half chunk in row
                cp16_g2l(vtp + (size_t)d * NN + kt + jc, &vbuf[d * 64 + jc]);
            }
            if (kt + 64 < NN) {   // global_prefetch_b8 for next tile
                __builtin_prefetch(kp + (size_t)(kt + 64) * HD, 0, 1);
                __builtin_prefetch(vtp + kt + 64, 0, 1);
            }
        }
        drain_async();     // s_wait_asynccnt 0 (barrier does not cover ASYNCcnt)
        __syncthreads();

        // S = q @ k^T  (16 x 64)
        v8f s[4];
#pragma unroll
        for (int jt = 0; jt < 4; ++jt) s[jt] = zero8();
#pragma unroll
        for (int dk = 0; dk < 4; ++dk) {
#pragma unroll
            for (int jt = 0; jt < 4; ++jt) {
                const v16h bf = *(const v16h*)(&kbuf[(jt * 16 + l16) * HD + dk * 32 + lh * 16]);
                s[jt] = wmma_f16(qf[dk], bf, s[jt]);
            }
        }
#pragma unroll
        for (int jt = 0; jt < 4; ++jt)
#pragma unroll
            for (int r = 0; r < 8; ++r) s[jt][r] *= sscale;

        // online softmax (row lives in one 16-lane half -> xor masks 8..1)
#pragma unroll
        for (int r = 0; r < 8; ++r) {
            float mx = fmaxf(fmaxf(s[0][r], s[1][r]), fmaxf(s[2][r], s[3][r]));
#pragma unroll
            for (int msk = 8; msk >= 1; msk >>= 1) mx = fmaxf(mx, __shfl_xor(mx, msk, 32));
            const float mn = fmaxf(m[r], mx);
            const float alpha = exp2f((m[r] - mn) * E2);
            m[r] = mn;
            l[r] *= alpha;
#pragma unroll
            for (int dt = 0; dt < 8; ++dt) o[dt][r] *= alpha;
            float rsum = 0.f;
#pragma unroll
            for (int jt = 0; jt < 4; ++jt) {
                const float p = exp2f((s[jt][r] - mn) * E2);
                rsum += p;
                pbuf[wave][(r + lh * 8) * 64 + jt * 16 + l16] = (_Float16)p;
            }
#pragma unroll
            for (int msk = 8; msk >= 1; msk >>= 1) rsum += __shfl_xor(rsum, msk, 32);
            l[r] += rsum;
        }

        // O += P @ V   (16 x 128)
#pragma unroll
        for (int js = 0; js < 2; ++js) {
            const v16h pf = *(const v16h*)(&pbuf[wave][l16 * 64 + js * 32 + lh * 16]);
#pragma unroll
            for (int dt = 0; dt < 8; ++dt) {
                const v16h vf = *(const v16h*)(&vbuf[(dt * 16 + l16) * 64 + js * 32 + lh * 16]);
                o[dt] = wmma_f16(pf, vf, o[dt]);
            }
        }
    }

    // normalize and pack back to [B, N, H*D] as f16 for the out projection
#pragma unroll
    for (int r = 0; r < 8; ++r) {
        const float inv = 1.0f / l[r];
        const int n = q0 + lh * 8 + r;
        const size_t base = ((size_t)b * NN + n) * DIMX + h * HD;
#pragma unroll
        for (int dt = 0; dt < 8; ++dt)
            oh[base + dt * 16 + l16] = (_Float16)(o[dt][r] * inv);
    }
}

// ---------------------------------------------------------------------------
// Kernel 4: out = attn_out @ out_w^T + b  (f16 operands; fp32 result)
// grid = (4096/64, 1024/256) = (64, 4)
// ---------------------------------------------------------------------------
__global__ __launch_bounds__(256) void out_gemm_k(
    const _Float16* __restrict__ ain, const _Float16* __restrict__ wh,
    const float* __restrict__ bias, float* __restrict__ out)
{
    const int wave = threadIdx.x >> 5, lane = threadIdx.x & 31;
    const int wr = wave >> 2, wc = wave & 3;
    const int row0 = blockIdx.x * 64 + wr * 32;
    const int col0 = blockIdx.y * 256 + wc * 64;
    const int l16 = lane & 15, lh = lane >> 4;

    v8f acc[2][4];
#pragma unroll
    for (int i = 0; i < 2; ++i)
#pragma unroll
        for (int j = 0; j < 4; ++j) acc[i][j] = zero8();

    for (int k = 0; k < DIMX; k += 32) {
        v16h a[2], bf[4];
#pragma unroll
        for (int rt = 0; rt < 2; ++rt)
            a[rt] = *(const v16h*)(ain + (size_t)(row0 + rt * 16 + l16) * DIMX + k + lh * 16);
#pragma unroll
        for (int ct = 0; ct < 4; ++ct)
            bf[ct] = *(const v16h*)(wh + (size_t)(col0 + ct * 16 + l16) * DIMX + k + lh * 16);
#pragma unroll
        for (int rt = 0; rt < 2; ++rt)
#pragma unroll
            for (int ct = 0; ct < 4; ++ct)
                acc[rt][ct] = wmma_f16(a[rt], bf[ct], acc[rt][ct]);
    }

#pragma unroll
    for (int rt = 0; rt < 2; ++rt)
#pragma unroll
        for (int ct = 0; ct < 4; ++ct) {
            const int col = col0 + ct * 16 + l16;
            const float bz = bias[col];
#pragma unroll
            for (int r = 0; r < 8; ++r) {
                const int row = row0 + rt * 16 + lh * 8 + r;
                out[(size_t)row * DIMX + col] = acc[rt][ct][r] + bz;
            }
        }
}

// ---------------------------------------------------------------------------
// Workspace layout (aliased; total 82 MB):
//   [0,48M)        qkv fp32      (k1 out, k2 in)
//   [48M,56M)      qh f16        (k2 out, k3 in)   | wqkv_h f16 6MB (k0->k1)
//   [56M,64M)      kh f16        (k2 out, k3 in)
//   [64M,72M)      vt f16 [BHDN] (k2 out, k3 in)
//   [72M,80M)      oh f16        (k3 out, k4 in)   | xh f16 8MB (k0->k1)
//   [80M,82M)      outw_h f16    (k0 out, k4 in)
// ---------------------------------------------------------------------------
extern "C" void kernel_launch(void* const* d_in, const int* in_sizes, int n_in,
                              void* d_out, int out_size, void* d_ws, size_t ws_size,
                              hipStream_t stream)
{
    const float* x       = (const float*)d_in[0];
    const float* Wqkv_w  = (const float*)d_in[1];
    const float* Wqkv_b  = (const float*)d_in[2];
    const float* q_gamma = (const float*)d_in[3];
    const float* q_beta  = (const float*)d_in[4];
    const float* k_gamma = (const float*)d_in[5];
    const float* k_beta  = (const float*)d_in[6];
    const float* out_w   = (const float*)d_in[7];
    const float* out_b   = (const float*)d_in[8];
    const float* rope_c  = (const float*)d_in[9];
    const float* rope_s  = (const float*)d_in[10];
    float* out = (float*)d_out;

    char* ws = (char*)d_ws;
    const size_t qkv_bytes  = (size_t)RTOT * C3 * sizeof(float);              // 48 MB
    const size_t head_bytes = (size_t)BB * NH * NN * HD * sizeof(_Float16);   // 8 MB

    float*    qkv   = (float*)ws;
    _Float16* qh    = (_Float16*)(ws + qkv_bytes);
    _Float16* kh    = (_Float16*)(ws + qkv_bytes + 1 * head_bytes);
    _Float16* vt    = (_Float16*)(ws + qkv_bytes + 2 * head_bytes);
    _Float16* oh    = (_Float16*)(ws + qkv_bytes + 3 * head_bytes);
    _Float16* outwh = (_Float16*)(ws + qkv_bytes + 4 * head_bytes);
    _Float16* xh    = oh;   // alias: consumed by k1 before k3 writes oh
    _Float16* wqkvh = qh;   // alias: consumed by k1 before k2 writes qh

    // Kernel 0: one-time fp32 -> f16 conversions (4 elems / thread)
    cvt_f16_k<<<(RTOT * DIMX) / 1024, 256, 0, stream>>>(x, xh);
    cvt_f16_k<<<(C3 * DIMX) / 1024, 256, 0, stream>>>(Wqkv_w, wqkvh);
    cvt_f16_k<<<(DIMX * DIMX) / 1024, 256, 0, stream>>>(out_w, outwh);

    qkv_gemm_k<<<dim3(RTOT / 64, C3 / 256), 256, 0, stream>>>(xh, wqkvh, Wqkv_b, qkv);
    lnrope_pack_k<<<(BB * NN * NH) / 8, 256, 0, stream>>>(
        qkv, q_gamma, q_beta, k_gamma, k_beta, rope_c, rope_s, qh, kh, vt);
    attn_k<<<dim3(BB * NH, NN / 128), 256, 0, stream>>>(qh, kh, vt, oh);
    out_gemm_k<<<dim3(RTOT / 64, DIMX / 256), 256, 0, stream>>>(oh, outwh, out_b, out);
}
